// ManualAttention_20950850470353
// MI455X (gfx1250) — matrix operand order repaired and checked by reading, not compiled, not run on hardware
//
#include <hip/hip_runtime.h>
#include <hip/hip_bf16.h>

typedef __attribute__((ext_vector_type(16))) __bf16 v16bf;
typedef __attribute__((ext_vector_type(8)))  __bf16 v8bf;
typedef __attribute__((ext_vector_type(8)))  float  v8f;
typedef int v4i_vs __attribute__((vector_size(16)));  // matches builtin param

#define AS1 __attribute__((address_space(1)))
#define AS3 __attribute__((address_space(3)))

// D = A(16x32 bf16) x B(32x16 bf16) + C(16x16 f32)
static __device__ __forceinline__ v8f wmma_bf16(v16bf a, v16bf b, v8f c) {
  return __builtin_amdgcn_wmma_f32_16x16x32_bf16(
      /*neg_a=*/false, a, /*neg_b=*/false, b,
      /*c_mod=*/(short)0, c, /*reuse_a=*/false, /*reuse_b=*/false);
}

static __device__ __forceinline__ v16bf combine16(v8bf lo, v8bf hi) {
  v16bf r;
#pragma unroll
  for (int i = 0; i < 8; ++i) { r[i] = lo[i]; r[i + 8] = hi[i]; }
  return r;
}

// 16B global -> LDS copy; async (ASYNCcnt) when the gfx1250 builtin exists.
static __device__ __forceinline__ void cp16_g2l(const __bf16* g, __bf16* l) {
#if __has_builtin(__builtin_amdgcn_global_load_async_to_lds_b128)
  __builtin_amdgcn_global_load_async_to_lds_b128(
      (AS1 v4i_vs*)(unsigned long long)(const void*)g,
      (AS3 v4i_vs*)(unsigned)(unsigned long long)(void*)l, 0, 0);
#else
  *reinterpret_cast<v8bf*>(l) = *reinterpret_cast<const v8bf*>(g);
#endif
}
static __device__ __forceinline__ void wait_async_all() {
#if __has_builtin(__builtin_amdgcn_s_wait_asynccnt)
  __builtin_amdgcn_s_wait_asynccnt(0);
#else
  asm volatile("s_wait_asynccnt 0" ::: "memory");
#endif
}

static __device__ __forceinline__ float red_max16(float v) {
  v = fmaxf(v, __shfl_xor(v, 1));
  v = fmaxf(v, __shfl_xor(v, 2));
  v = fmaxf(v, __shfl_xor(v, 4));
  v = fmaxf(v, __shfl_xor(v, 8));
  return v;
}
static __device__ __forceinline__ float red_sum16(float v) {
  v += __shfl_xor(v, 1);
  v += __shfl_xor(v, 2);
  v += __shfl_xor(v, 4);
  v += __shfl_xor(v, 8);
  return v;
}

#define D_MODEL 1024
#define N_SEQ   2048
#define N_HEADS 16
#define D_HEAD  64

// ---------------- GEMM tiling --------------------------------------------
// Block tile 128x128, BK=32, 8 waves as 4x2; wave tile 32x64 = 2x4 WMMA.
#define BM 128
#define BN 128
#define BK 32
#define LDA 40   // bf16 row stride: 64B data + 16B pad (16B aligned)
#define LDB 40

// ---------------- Kernel 1: QKV projection -------------------------------
// C[M=8192, E=3072] = x @ w_qkv^T + b_qkv, scattered to per-head bf16 Q/K/V.
__global__ __launch_bounds__(256) void qkv_gemm_kernel(
    const float* __restrict__ x, const float* __restrict__ w,
    const float* __restrict__ bias,
    __bf16* __restrict__ qb, __bf16* __restrict__ kb, __bf16* __restrict__ vb) {
  __shared__ __align__(16) __bf16 lds_a[BM * LDA];  // [m][k]
  __shared__ __align__(16) __bf16 lds_b[BN * LDB];  // [e][k]
  const int tid = threadIdx.x;
  const int lane = tid & 31, wid = tid >> 5;
  const int hf = lane >> 4, lp = lane & 15;
  const int wr = wid >> 1, wc = wid & 1;
  const int m0 = blockIdx.x * BM;
  const int e0 = blockIdx.y * BN;

  v8f acc[2][4] = {};
  for (int kt = 0; kt < D_MODEL; kt += BK) {
    // stage A (fp32 -> bf16): 1024 float4 chunks, 4 per thread
#pragma unroll
    for (int i = 0; i < 4; ++i) {
      int c = tid + 256 * i;
      int row = c >> 3, off = (c & 7) * 4;
      const float4 f = *reinterpret_cast<const float4*>(
          &x[(size_t)(m0 + row) * D_MODEL + kt + off]);
      __bf16* dst = &lds_a[row * LDA + off];
      dst[0] = (__bf16)f.x; dst[1] = (__bf16)f.y;
      dst[2] = (__bf16)f.z; dst[3] = (__bf16)f.w;
    }
    // stage B (fp32 -> bf16): 1024 float4 chunks, 4 per thread
#pragma unroll
    for (int i = 0; i < 4; ++i) {
      int c = tid + 256 * i;
      int row = c >> 3, off = (c & 7) * 4;
      const float4 f = *reinterpret_cast<const float4*>(
          &w[(size_t)(e0 + row) * D_MODEL + kt + off]);
      __bf16* dst = &lds_b[row * LDB + off];
      dst[0] = (__bf16)f.x; dst[1] = (__bf16)f.y;
      dst[2] = (__bf16)f.z; dst[3] = (__bf16)f.w;
    }
    __syncthreads();
    v16bf afrag[2], bfrag[4];
#pragma unroll
    for (int t = 0; t < 2; ++t) {  // A layout: lane m = lane&15; k halves 8h.., 16+8h..
      int row = wr * 32 + t * 16 + lp;
      v8bf lo = *reinterpret_cast<const v8bf*>(&lds_a[row * LDA + 8 * hf]);
      v8bf hi = *reinterpret_cast<const v8bf*>(&lds_a[row * LDA + 16 + 8 * hf]);
      afrag[t] = combine16(lo, hi);
    }
#pragma unroll
    for (int t = 0; t < 4; ++t) {  // B layout: lane n = lane&15; k halves 8h.., 16+8h..
      int col = wc * 64 + t * 16 + lp;
      v8bf lo = *reinterpret_cast<const v8bf*>(&lds_b[col * LDB + 8 * hf]);
      v8bf hi = *reinterpret_cast<const v8bf*>(&lds_b[col * LDB + 16 + 8 * hf]);
      bfrag[t] = combine16(lo, hi);
    }
#pragma unroll
    for (int i = 0; i < 2; ++i)
#pragma unroll
      for (int j = 0; j < 4; ++j)
        acc[i][j] = wmma_bf16(afrag[i], bfrag[j], acc[i][j]);
    __syncthreads();
  }
  // epilogue: e block (128 wide) lies within one `which` region; head varies
  const int which = e0 >> 10;
  __bf16* dst = (which == 0) ? qb : ((which == 1) ? kb : vb);
#pragma unroll
  for (int i = 0; i < 2; ++i)
#pragma unroll
    for (int j = 0; j < 4; ++j)
#pragma unroll
      for (int r = 0; r < 8; ++r) {  // C layout: row = r + 8h, col = lane&15
        int mg = m0 + wr * 32 + i * 16 + r + 8 * hf;
        int e = e0 + wc * 64 + j * 16 + lp;
        float v = acc[i][j][r] + bias[e];
        int head = (e >> 6) & (N_HEADS - 1);
        int bidx = mg >> 11, n = mg & (N_SEQ - 1);
        dst[(((size_t)(bidx * N_HEADS + head)) * N_SEQ + n) * D_HEAD + (e & 63)] =
            (__bf16)v;
      }
}

// ---------------- Kernel 2: flash attention ------------------------------
#define BQ  128
#define BKV 64
#define LKK 72  // lds_k row stride (bf16): 128B data + 16B pad
#define LVT 72  // lds_vt row stride: 64 kv cols + 8 pad
#define LPP 72  // per-wave P patch row stride: 64 cols + 8 pad

__global__ __launch_bounds__(256) void attn_kernel(
    const __bf16* __restrict__ qb, const __bf16* __restrict__ kb,
    const __bf16* __restrict__ vb, __bf16* __restrict__ ab) {
  __shared__ __align__(16) __bf16 lds_k[BKV * LKK];       // [kvrow][d]
  __shared__ __align__(16) __bf16 lds_vt[D_HEAD * LVT];   // [d][kvrow] (transposed V)
  __shared__ __align__(16) __bf16 lds_p[8 * 16 * LPP];    // per wave: [m][n]
  const int tid = threadIdx.x;
  const int lane = tid & 31, wid = tid >> 5;
  const int hf = lane >> 4, lp = lane & 15;
  const int qt = blockIdx.x & 15;  // N/BQ = 16
  const int bh = blockIdx.x >> 4;  // 0..63 = b*H+h
  const size_t base = (size_t)bh * N_SEQ * D_HEAD;

  // Q fragments for this wave's 16 query rows: contiguous 16B reads per lane
  v16bf qfrag[2];
  {
    int row = qt * BQ + wid * 16 + lp;
    const __bf16* qp = qb + base + (size_t)row * D_HEAD;
#pragma unroll
    for (int ks = 0; ks < 2; ++ks) {
      v8bf lo = *reinterpret_cast<const v8bf*>(qp + 32 * ks + 8 * hf);
      v8bf hi = *reinterpret_cast<const v8bf*>(qp + 32 * ks + 16 + 8 * hf);
      qfrag[ks] = combine16(lo, hi);
    }
  }
  float mrow[8], lrow[8];
#pragma unroll
  for (int r = 0; r < 8; ++r) { mrow[r] = -3.0e38f; lrow[r] = 0.f; }
  v8f o[4] = {};

  for (int t = 0; t < N_SEQ / BKV; ++t) {
    // stage K row-major (async copy) and V transposed (manual)
#pragma unroll
    for (int i = 0; i < 2; ++i) {
      int c = tid + 256 * i;            // 512 chunks of 8 bf16
      int row = c >> 3, off = (c & 7) * 8;
      const size_t g = base + (size_t)(t * BKV + row) * D_HEAD + off;
      cp16_g2l(kb + g, &lds_k[row * LKK + off]);
      v8bf vv = *reinterpret_cast<const v8bf*>(vb + g);
#pragma unroll
      for (int j = 0; j < 8; ++j) lds_vt[(off + j) * LVT + row] = vv[j];
    }
    if (t + 1 < N_SEQ / BKV) {  // prefetch next tile -> global_prefetch_b8
      const size_t g = base + (size_t)((t + 1) * BKV + (tid >> 2)) * D_HEAD;
      __builtin_prefetch(kb + g, 0, 1);
      __builtin_prefetch(vb + g, 0, 1);
    }
    wait_async_all();
    __syncthreads();
    // scores S = Q K^T / sqrt(dh): four 16-col tiles, dh=64 -> 2 k-steps each
    v8f s[4];
#pragma unroll
    for (int ct = 0; ct < 4; ++ct) {
      v8f z = {};
      s[ct] = z;
#pragma unroll
      for (int ks = 0; ks < 2; ++ks) {
        int krow = ct * 16 + lp;
        v8bf lo = *reinterpret_cast<const v8bf*>(
            &lds_k[krow * LKK + 32 * ks + 8 * hf]);
        v8bf hi = *reinterpret_cast<const v8bf*>(
            &lds_k[krow * LKK + 32 * ks + 16 + 8 * hf]);
        s[ct] = wmma_bf16(qfrag[ks], combine16(lo, hi), s[ct]);
      }
#pragma unroll
      for (int r = 0; r < 8; ++r) s[ct][r] *= 0.125f;  // 1/sqrt(64)
    }
    // online softmax (rows live in 16-lane halves; r indexes 8 rows/lane)
    float corr[8];
#pragma unroll
    for (int r = 0; r < 8; ++r) {
      float v = fmaxf(fmaxf(s[0][r], s[1][r]), fmaxf(s[2][r], s[3][r]));
      v = red_max16(v);
      float mn = fmaxf(mrow[r], v);
      float c = __expf(mrow[r] - mn);
      float p0 = __expf(s[0][r] - mn);
      float p1 = __expf(s[1][r] - mn);
      float p2 = __expf(s[2][r] - mn);
      float p3 = __expf(s[3][r] - mn);
      s[0][r] = p0; s[1][r] = p1; s[2][r] = p2; s[3][r] = p3;
      lrow[r] = lrow[r] * c + red_sum16((p0 + p1) + (p2 + p3));
      mrow[r] = mn;
      corr[r] = c;
    }
#pragma unroll
    for (int j = 0; j < 4; ++j)
#pragma unroll
      for (int r = 0; r < 8; ++r) o[j][r] *= corr[r];
    // C-layout -> A-layout transpose of P[16x64] through per-wave LDS patch
    __bf16* pw = &lds_p[wid * 16 * LPP];
#pragma unroll
    for (int r = 0; r < 8; ++r) {
      int m = r + 8 * hf;
#pragma unroll
      for (int ct = 0; ct < 4; ++ct)
        pw[m * LPP + ct * 16 + lp] = (__bf16)s[ct][r];
    }
    asm volatile("s_wait_dscnt 0" ::: "memory");  // intra-wave LDS RAW
    v16bf pfrag[2];
#pragma unroll
    for (int ks = 0; ks < 2; ++ks) {
      v8bf plo = *reinterpret_cast<const v8bf*>(
          &pw[lp * LPP + 32 * ks + 8 * hf]);
      v8bf phi = *reinterpret_cast<const v8bf*>(
          &pw[lp * LPP + 32 * ks + 16 + 8 * hf]);
      pfrag[ks] = combine16(plo, phi);
    }
    // O += P V : contraction over 64 kv rows (2 k-steps), 4 d-tiles
#pragma unroll
    for (int j = 0; j < 4; ++j) {
      int d = j * 16 + lp;
#pragma unroll
      for (int ks = 0; ks < 2; ++ks) {
        v8bf lo = *reinterpret_cast<const v8bf*>(
            &lds_vt[d * LVT + 32 * ks + 8 * hf]);
        v8bf hi = *reinterpret_cast<const v8bf*>(
            &lds_vt[d * LVT + 32 * ks + 16 + 8 * hf]);
        o[j] = wmma_bf16(pfrag[ks], combine16(lo, hi), o[j]);
      }
    }
    __syncthreads();
  }
  // normalize and write [B, N, H*dh] bf16 for final projection
  const int b = bh >> 4, head = bh & (N_HEADS - 1);
#pragma unroll
  for (int r = 0; r < 8; ++r) {
    float inv = 1.0f / lrow[r];
    int row = qt * BQ + wid * 16 + r + 8 * hf;
    size_t rb = ((size_t)b * N_SEQ + row) * D_MODEL + head * D_HEAD;
#pragma unroll
    for (int j = 0; j < 4; ++j) ab[rb + j * 16 + lp] = (__bf16)(o[j][r] * inv);
  }
}

// ---------------- Kernel 3: output projection ----------------------------
__global__ __launch_bounds__(256) void out_gemm_kernel(
    const __bf16* __restrict__ a, const float* __restrict__ w,
    const float* __restrict__ bias, float* __restrict__ out) {
  __shared__ __align__(16) __bf16 lds_a[BM * LDA];
  __shared__ __align__(16) __bf16 lds_b[BN * LDB];
  const int tid = threadIdx.x;
  const int lane = tid & 31, wid = tid >> 5;
  const int hf = lane >> 4, lp = lane & 15;
  const int wr = wid >> 1, wc = wid & 1;
  const int m0 = blockIdx.x * BM;
  const int e0 = blockIdx.y * BN;

  v8f acc[2][4] = {};
  for (int kt = 0; kt < D_MODEL; kt += BK) {
    // stage A (already bf16, async copy): 512 16B chunks, 2 per thread
#pragma unroll
    for (int i = 0; i < 2; ++i) {
      int c = tid + 256 * i;
      int row = c >> 2, off = (c & 3) * 8;
      cp16_g2l(&a[(size_t)(m0 + row) * D_MODEL + kt + off],
               &lds_a[row * LDA + off]);
    }
    // stage B (fp32 -> bf16): 1024 float4 chunks, 4 per thread
#pragma unroll
    for (int i = 0; i < 4; ++i) {
      int c = tid + 256 * i;
      int row = c >> 3, off = (c & 7) * 4;
      const float4 f = *reinterpret_cast<const float4*>(
          &w[(size_t)(e0 + row) * D_MODEL + kt + off]);
      __bf16* dst = &lds_b[row * LDB + off];
      dst[0] = (__bf16)f.x; dst[1] = (__bf16)f.y;
      dst[2] = (__bf16)f.z; dst[3] = (__bf16)f.w;
    }
    wait_async_all();
    __syncthreads();
    v16bf afrag[2], bfrag[4];
#pragma unroll
    for (int t = 0; t < 2; ++t) {
      int row = wr * 32 + t * 16 + lp;
      v8bf lo = *reinterpret_cast<const v8bf*>(&lds_a[row * LDA + 8 * hf]);
      v8bf hi = *reinterpret_cast<const v8bf*>(&lds_a[row * LDA + 16 + 8 * hf]);
      afrag[t] = combine16(lo, hi);
    }
#pragma unroll
    for (int t = 0; t < 4; ++t) {
      int col = wc * 64 + t * 16 + lp;
      v8bf lo = *reinterpret_cast<const v8bf*>(&lds_b[col * LDB + 8 * hf]);
      v8bf hi = *reinterpret_cast<const v8bf*>(&lds_b[col * LDB + 16 + 8 * hf]);
      bfrag[t] = combine16(lo, hi);
    }
#pragma unroll
    for (int i = 0; i < 2; ++i)
#pragma unroll
      for (int j = 0; j < 4; ++j)
        acc[i][j] = wmma_bf16(afrag[i], bfrag[j], acc[i][j]);
    __syncthreads();
  }
#pragma unroll
  for (int i = 0; i < 2; ++i)
#pragma unroll
    for (int j = 0; j < 4; ++j)
#pragma unroll
      for (int r = 0; r < 8; ++r) {
        int mg = m0 + wr * 32 + i * 16 + r + 8 * hf;
        int e = e0 + wc * 64 + j * 16 + lp;
        out[(size_t)mg * D_MODEL + e] = acc[i][j][r] + bias[e];
      }
}

// ---------------- launch -------------------------------------------------
extern "C" void kernel_launch(void* const* d_in, const int* in_sizes, int n_in,
                              void* d_out, int out_size, void* d_ws, size_t ws_size,
                              hipStream_t stream) {
  const float* x = (const float*)d_in[0];
  const float* w_qkv = (const float*)d_in[1];
  const float* b_qkv = (const float*)d_in[2];
  const float* w_out = (const float*)d_in[3];
  const float* b_out = (const float*)d_in[4];
  float* out = (float*)d_out;

  // workspace: 4 bf16 buffers of B*N*D = 8M elems (16 MB) each = 64 MB
  char* wsb = (char*)d_ws;
  const size_t seg = (size_t)4 * N_SEQ * D_MODEL * sizeof(__bf16);
  __bf16* qb = (__bf16*)(wsb);
  __bf16* kb = (__bf16*)(wsb + seg);
  __bf16* vb = (__bf16*)(wsb + 2 * seg);
  __bf16* ab = (__bf16*)(wsb + 3 * seg);

  dim3 blk(256);
  // M=8192 rows, E=3072 cols
  qkv_gemm_kernel<<<dim3(8192 / BM, 3 * D_MODEL / BN), blk, 0, stream>>>(
      x, w_qkv, b_qkv, qb, kb, vb);
  // 4*16 (b,h) pairs x 16 query tiles
  attn_kernel<<<dim3(64 * (N_SEQ / BQ)), blk, 0, stream>>>(qb, kb, vb, ab);
  // M=8192 rows, E=1024 cols
  out_gemm_kernel<<<dim3(8192 / BM, D_MODEL / BN), blk, 0, stream>>>(
      ab, w_out, b_out, out);
}
